// GVAE_84035330113721
// MI455X (gfx1250) — matrix operand order repaired and checked by reading, not compiled, not run on hardware
//
#include <hip/hip_runtime.h>
#include <hip/hip_bf16.h>

// ---------------------------------------------------------------------------
// GVAE forward for MI455X (gfx1250, wave32, WMMA).
// Dense layers -> v_wmma_f32_16x16x32_f16 (f16 in, f32 accum), b128 loads,
// 64x64 register-blocked tiles (16 WMMA per k-step, 2 b128 loads per WMMA).
// Sparse graph ops -> f32 VALU + global atomics + LDS reductions.
// ---------------------------------------------------------------------------

#define NN 65536
#define EE 262144
#define BB 8192
#define DD 64
#define HH 4
#define NPG 8           // nodes per graph
#define ATOM_OUT 1380
#define EDGE_OUT 210

typedef __attribute__((ext_vector_type(16))) _Float16 v16h;
typedef __attribute__((ext_vector_type(8)))  float    v8f;

// -------------------------- small device helpers ---------------------------
__device__ __forceinline__ unsigned enc_f32(float x) {
    unsigned u = __float_as_uint(x);
    return (u & 0x80000000u) ? ~u : (u | 0x80000000u);
}
__device__ __forceinline__ float dec_f32(unsigned u) {
    u = (u & 0x80000000u) ? (u & 0x7fffffffu) : ~u;
    return __uint_as_float(u);
}
__device__ __forceinline__ float sigf(float x) { return 1.0f / (1.0f + __expf(-x)); }

__device__ __forceinline__ v16h cvt16(float4 f0, float4 f1, float4 f2, float4 f3)
{
    v16h r;
    r[0]  = (_Float16)f0.x; r[1]  = (_Float16)f0.y;
    r[2]  = (_Float16)f0.z; r[3]  = (_Float16)f0.w;
    r[4]  = (_Float16)f1.x; r[5]  = (_Float16)f1.y;
    r[6]  = (_Float16)f1.z; r[7]  = (_Float16)f1.w;
    r[8]  = (_Float16)f2.x; r[9]  = (_Float16)f2.y;
    r[10] = (_Float16)f2.z; r[11] = (_Float16)f2.w;
    r[12] = (_Float16)f3.x; r[13] = (_Float16)f3.y;
    r[14] = (_Float16)f3.z; r[15] = (_Float16)f3.w;
    return r;
}
// A fragment (16x32 f16): lane already offset by hi*8; runs [0..7] and [16..23]
__device__ __forceinline__ v16h load_fragA(const float* p)
{
    return cvt16(*(const float4*)(p), *(const float4*)(p + 4),
                 *(const float4*)(p + 16), *(const float4*)(p + 20));
}
// B fragment (32x16 f16): lane already offset by hi*8; runs [0..7] and [16..23]
__device__ __forceinline__ v16h load_fragB(const float* p)
{
    return cvt16(*(const float4*)(p), *(const float4*)(p + 4),
                 *(const float4*)(p + 16), *(const float4*)(p + 20));
}

// ---------------------------------------------------------------------------
// 64x64-tile WMMA GEMM:  Y[M,Nout] = X @ W^T (+bias)(+=Y)(relu)
// Requires K%32==0, M%64==0, covered cols %64==0. One wave -> 4x4 WMMA tiles.
// flags: bit0 = accumulate into Y, bit1 = relu
// ---------------------------------------------------------------------------
__global__ __launch_bounds__(32)
void gemm_wmma64_kernel(const float* __restrict__ X, int ldx,
                        const float* __restrict__ W,
                        const float* __restrict__ bias,
                        float* __restrict__ Y, int ldy, int K, int flags)
{
    const int l    = threadIdx.x;
    const int row0 = blockIdx.x << 6;
    const int col0 = blockIdx.y << 6;
    const int ml   = l & 15;
    const int hi   = l >> 4;

    const float* xrow[4];
    const float* wrow[4];
#pragma unroll
    for (int mi = 0; mi < 4; ++mi)
        xrow[mi] = X + (size_t)(row0 + mi * 16 + ml) * ldx + (hi << 3);
#pragma unroll
    for (int ni = 0; ni < 4; ++ni)
        wrow[ni] = W + (size_t)(col0 + ni * 16 + ml) * K + (hi << 3);

    v8f acc[4][4] = {};
    for (int kt = 0; kt < K; kt += 32) {
        v16h a[4], b[4];
#pragma unroll
        for (int mi = 0; mi < 4; ++mi) a[mi] = load_fragA(xrow[mi] + kt);
#pragma unroll
        for (int ni = 0; ni < 4; ++ni) b[ni] = load_fragB(wrow[ni] + kt);
#pragma unroll
        for (int mi = 0; mi < 4; ++mi)
#pragma unroll
            for (int ni = 0; ni < 4; ++ni)
                acc[mi][ni] = __builtin_amdgcn_wmma_f32_16x16x32_f16(
                    false, a[mi], false, b[ni], (short)0, acc[mi][ni], false, false);
    }

#pragma unroll
    for (int ni = 0; ni < 4; ++ni) {
        int n = col0 + ni * 16 + ml;
        float bv = bias ? bias[n] : 0.0f;
#pragma unroll
        for (int mi = 0; mi < 4; ++mi) {
#pragma unroll
            for (int r = 0; r < 8; ++r) {
                int m = row0 + mi * 16 + r + (hi << 3);
                size_t idx = (size_t)m * ldy + n;
                float y = acc[mi][ni][r] + bv;
                if (flags & 1) y += Y[idx];
                if (flags & 2) y = fmaxf(y, 0.0f);
                Y[idx] = y;
            }
        }
    }
}

// ---------------------------------------------------------------------------
// 16x16-tile WMMA GEMM for ragged column remainders (cols colbase..Nout).
// Clamped loads, guarded stores. Requires K%32==0, M%16==0.
// ---------------------------------------------------------------------------
__global__ __launch_bounds__(32)
void gemm_wmma16_kernel(const float* __restrict__ X, int ldx,
                        const float* __restrict__ W,
                        const float* __restrict__ bias,
                        float* __restrict__ Y, int ldy,
                        int colbase, int Nout, int K, int flags)
{
    const int l    = threadIdx.x;
    const int row0 = blockIdx.x << 4;
    const int col0 = colbase + (blockIdx.y << 4);
    const int ml   = l & 15;
    const int hi   = l >> 4;
    const int n    = col0 + ml;
    const int nc   = (n < Nout) ? n : (Nout - 1);

    const float* xrow = X + (size_t)(row0 + ml) * ldx + (hi << 3);
    const float* wrow = W + (size_t)nc * K + (hi << 3);

    v8f c = {};
#pragma unroll 2
    for (int kt = 0; kt < K; kt += 32) {
        v16h a = load_fragA(xrow + kt);
        v16h b = load_fragB(wrow + kt);
        c = __builtin_amdgcn_wmma_f32_16x16x32_f16(false, a, false, b,
                                                   (short)0, c, false, false);
    }
    if (n >= Nout) return;
    float bv = bias ? bias[n] : 0.0f;
#pragma unroll
    for (int r = 0; r < 8; ++r) {
        int m = r + (hi << 3);
        size_t idx = (size_t)(row0 + m) * ldy + n;
        float y = c[r] + bv;
        if (flags & 1) y += Y[idx];
        if (flags & 2) y = fmaxf(y, 0.0f);
        Y[idx] = y;
    }
}

// Zero-pad [rows,4] -> [rows,32] so conv1 GEMMs can use the unguarded path.
__global__ __launch_bounds__(256)
void pad4to32_kernel(const float* __restrict__ src, float* __restrict__ dst, int rows)
{
    int t = blockIdx.x * blockDim.x + threadIdx.x;
    if (t >= rows * 32) return;
    int r = t >> 5, cidx = t & 31;
    dst[t] = (cidx < 4) ? src[r * 4 + cidx] : 0.0f;
}

// ---------------------------------------------------------------------------
// Edge attention: logits + running segment max (encoded uint atomicMax)
// ---------------------------------------------------------------------------
__global__ __launch_bounds__(256)
void edge_logits_kernel(const float* __restrict__ q, const float* __restrict__ k,
                        const int* __restrict__ src, const int* __restrict__ dst,
                        float* __restrict__ logits, unsigned* __restrict__ mbuf)
{
    int t = blockIdx.x * blockDim.x + threadIdx.x;
    if (t >= EE * HH) return;
    int e = t >> 2, h = t & 3;
    int s = src[e], d = dst[e];
    const float4* qp = (const float4*)(q + (size_t)d * (HH * DD) + h * DD);
    const float4* kp = (const float4*)(k + (size_t)s * (HH * DD) + h * DD);
    float acc = 0.0f;
#pragma unroll
    for (int i = 0; i < DD / 4; ++i) {
        float4 a = qp[i], b = kp[i];
        acc += a.x * b.x + a.y * b.y + a.z * b.z + a.w * b.w;
    }
    acc *= 0.125f;                       // 1/sqrt(64)
    logits[t] = acc;
    atomicMax(&mbuf[(size_t)d * HH + h], enc_f32(acc));
}

__global__ __launch_bounds__(256)
void edge_exp_kernel(float* __restrict__ logits, const unsigned* __restrict__ mbuf,
                     const int* __restrict__ dst, float* __restrict__ sbuf)
{
    int t = blockIdx.x * blockDim.x + threadIdx.x;
    if (t >= EE * HH) return;
    int e = t >> 2, h = t & 3;
    int d = dst[e];
    float a = __expf(logits[t] - dec_f32(mbuf[(size_t)d * HH + h]));
    logits[t] = a;                        // in-place -> alpha numerator
    atomicAdd(&sbuf[(size_t)d * HH + h], a);
}

__global__ __launch_bounds__(256)
void edge_aggr_kernel(const float* __restrict__ alpha, const float* __restrict__ sbuf,
                      const float* __restrict__ v,
                      const int* __restrict__ src, const int* __restrict__ dst,
                      float* __restrict__ out)
{
    int t = blockIdx.x * blockDim.x + threadIdx.x;
    if (t >= EE * DD) return;
    int e = t >> 6, c = t & 63;
    int s = src[e], d = dst[e];
    float acc = 0.0f;
#pragma unroll
    for (int h = 0; h < HH; ++h) {
        float w = alpha[(size_t)e * HH + h] / (sbuf[(size_t)d * HH + h] + 1e-16f);
        acc += w * v[(size_t)s * (HH * DD) + h * DD + c];
    }
    atomicAdd(&out[(size_t)d * DD + c], acc * 0.25f);   // head mean folded in
}

// ---------------------------------------------------------------------------
// Node epilogue: beta gate (LDS reduce over 64 ch) + relu -> hout
// block = 256 threads = 4 nodes x 64 channels
// ---------------------------------------------------------------------------
__global__ __launch_bounds__(256)
void node_epilogue_kernel(const float* __restrict__ out, const float* __restrict__ skip,
                          const float* __restrict__ betaW, float* __restrict__ hout)
{
    __shared__ float red[4][64];
    int c = threadIdx.x & 63;
    int j = threadIdx.x >> 6;
    int n = (blockIdx.x << 2) + j;
    float o = out[(size_t)n * DD + c];
    float s = skip[(size_t)n * DD + c];
    red[j][c] = betaW[c] * o + betaW[64 + c] * s + betaW[128 + c] * (o - s);
    __syncthreads();
#pragma unroll
    for (int off = 32; off > 0; off >>= 1) {
        if (c < off) red[j][c] += red[j][c + off];
        __syncthreads();
    }
    float beta = sigf(red[j][0]);
    float r = beta * s + (1.0f - beta) * o;
    hout[(size_t)n * DD + c] = fmaxf(r, 0.0f);
}

// ---------------------------------------------------------------------------
// BatchNorm (training stats): per-channel block reduction, then apply
// ---------------------------------------------------------------------------
__global__ __launch_bounds__(256)
void bn_stats_kernel(const float* __restrict__ h, float* __restrict__ stat)
{
    int c = blockIdx.x;                  // channel 0..63
    __shared__ float s1[256], s2[256];
    float a = 0.0f, b = 0.0f;
    for (int n = threadIdx.x; n < NN; n += 256) {
        float x = h[(size_t)n * DD + c];
        a += x; b += x * x;
    }
    s1[threadIdx.x] = a; s2[threadIdx.x] = b;
    __syncthreads();
#pragma unroll
    for (int off = 128; off > 0; off >>= 1) {
        if (threadIdx.x < off) {
            s1[threadIdx.x] += s1[threadIdx.x + off];
            s2[threadIdx.x] += s2[threadIdx.x + off];
        }
        __syncthreads();
    }
    if (threadIdx.x == 0) {
        float mu  = s1[0] * (1.0f / NN);
        float var = s2[0] * (1.0f / NN) - mu * mu;
        stat[c]      = mu;
        stat[64 + c] = rsqrtf(var + 1e-5f);
    }
}

__global__ __launch_bounds__(256)
void bn_apply_kernel(float* __restrict__ h, const float* __restrict__ stat,
                     const float* __restrict__ w, const float* __restrict__ b)
{
    int t = blockIdx.x * blockDim.x + threadIdx.x;
    if (t >= NN * DD) return;
    int c = t & 63;
    h[t] = (h[t] - stat[c]) * stat[64 + c] * w[c] + b[c];
}

// ---------------------------------------------------------------------------
// Set2Set pieces
// ---------------------------------------------------------------------------
__global__ __launch_bounds__(256)
void lstm_pointwise_kernel(const float* __restrict__ gates, float* __restrict__ hh,
                           float* __restrict__ cc, float* __restrict__ qstar)
{
    int t = blockIdx.x * blockDim.x + threadIdx.x;
    if (t >= BB * DD) return;
    int b = t >> 6, d = t & 63;
    const float* g = gates + (size_t)b * 256;
    float cv = sigf(g[64 + d]) * cc[t] + sigf(g[d]) * tanhf(g[128 + d]);
    float hv = sigf(g[192 + d]) * tanhf(cv);
    cc[t] = cv; hh[t] = hv;
    qstar[(size_t)b * 128 + d] = hv;     // first half of q_star
}

__global__ __launch_bounds__(256)
void s2s_attn_kernel(const float* __restrict__ h, const float* __restrict__ hh,
                     float* __restrict__ qstar)
{
    int g = blockIdx.x * blockDim.x + threadIdx.x;
    if (g >= BB) return;
    const float* hp = h  + (size_t)g * NPG * DD;
    const float* hq = hh + (size_t)g * DD;
    float e[NPG];
    float mx = -3.0e38f;
#pragma unroll
    for (int n = 0; n < NPG; ++n) {
        float acc = 0.0f;
        const float4* a4 = (const float4*)(hp + n * DD);
        const float4* q4 = (const float4*)hq;
        for (int i = 0; i < DD / 4; ++i) {
            float4 a = a4[i], q = q4[i];
            acc += a.x * q.x + a.y * q.y + a.z * q.z + a.w * q.w;
        }
        e[n] = acc;
        mx = fmaxf(mx, acc);
    }
    float sum = 0.0f;
#pragma unroll
    for (int n = 0; n < NPG; ++n) { e[n] = __expf(e[n] - mx); sum += e[n]; }
    float inv = 1.0f / (sum + 1e-16f);
    float* r = qstar + (size_t)g * 128 + 64;
    for (int c = 0; c < DD; ++c) {
        float acc = 0.0f;
#pragma unroll
        for (int n = 0; n < NPG; ++n) acc += e[n] * hp[n * DD + c];
        r[c] = acc * inv;
    }
}

__global__ __launch_bounds__(256)
void shear_fix_kernel(float* __restrict__ gbuf, const float* __restrict__ shear)
{
    int b = blockIdx.x * blockDim.x + threadIdx.x;
    if (b < BB) gbuf[(size_t)b * 128 + 127] = shear[b];
}

// ---------------------------------------------------------------------------
// Host driver
// ---------------------------------------------------------------------------
static inline void gemm(hipStream_t s, const float* X, int ldx, const float* W,
                        const float* bias, float* Y, int ldy,
                        int M, int Nout, int K, int flags)
{
    int n64 = Nout & ~63;
    if (n64) {
        dim3 grid(M >> 6, n64 >> 6);
        gemm_wmma64_kernel<<<grid, 32, 0, s>>>(X, ldx, W, bias, Y, ldy, K, flags);
    }
    if (Nout > n64) {
        dim3 grid(M >> 4, (Nout - n64 + 15) >> 4);
        gemm_wmma16_kernel<<<grid, 32, 0, s>>>(X, ldx, W, bias, Y, ldy,
                                               n64, Nout, K, flags);
    }
}

// Flattened param order (dict insertion order, depth-first):
//  0:x  1:edge_index  2:shear  3:batch_index
//  conv_i (i=0..3) at 4+9i: qW,qb,kW,kb,vW,vb,skipW,skipb,betaW
//  40..45: bn1.w,bn1.b,bn2.w,bn2.b,bn3.w,bn3.b
//  46..49: lstm Wih,Whh,bih,bhh
//  50,51: inter W,b   52,53: mu   54,55: logvar
//  56,57: dec1  58,59: dec2  60,61: atom  62,63: edge
#define CONV(i) (4 + 9 * (i))

extern "C" void kernel_launch(void* const* d_in, const int* in_sizes, int n_in,
                              void* d_out, int out_size, void* d_ws, size_t ws_size,
                              hipStream_t stream)
{
    (void)in_sizes; (void)n_in; (void)out_size; (void)ws_size;
    const float* x     = (const float*)d_in[0];
    const int*   src   = (const int*)d_in[1];          // edge_index[0]
    const int*   dst   = src + EE;                     // edge_index[1]
    const float* shear = (const float*)d_in[2];
    float* out = (float*)d_out;

    // ---- workspace layout (floats) ----
    float* ws = (float*)d_ws;
    size_t o = 0;
    float*    hA   = ws + o; o += (size_t)NN * DD;
    float*    hB   = ws + o; o += (size_t)NN * DD;
    float*    Q    = ws + o; o += (size_t)NN * HH * DD;
    float*    Kb   = ws + o; o += (size_t)NN * HH * DD;
    float*    V    = ws + o; o += (size_t)NN * HH * DD;
    float*    S    = ws + o; o += (size_t)NN * DD;
    float*    OUT  = ws + o; o += (size_t)NN * DD;
    float*    AL   = ws + o; o += (size_t)EE * HH;
    unsigned* MB   = (unsigned*)(ws + o); o += (size_t)NN * HH;
    float*    SB   = ws + o; o += (size_t)NN * HH;
    float*    ST   = ws + o; o += 128;
    float*    Xp   = ws + o; o += (size_t)NN * 32;      // conv1 padded input
    float*    Wp   = ws + o; o += (size_t)(256 * 3 + 64) * 32;  // conv1 padded weights
    // Set2Set / decoder buffers aliased over Q..V (dead after conv layers)
    float* qstar = Q;
    float* hh    = qstar + (size_t)BB * 128;
    float* cc    = hh    + (size_t)BB * DD;
    float* gates = cc    + (size_t)BB * DD;
    float* gbuf  = gates + (size_t)BB * 256;
    float* d1    = gbuf  + (size_t)BB * 128;
    float* d2    = d1    + (size_t)BB * 512;

    // ---- conv1 K=4 -> K=32 zero-padding ----
    float* Wq32 = Wp;
    float* Wk32 = Wq32 + 256 * 32;
    float* Wv32 = Wk32 + 256 * 32;
    float* Ws32 = Wv32 + 256 * 32;
    pad4to32_kernel<<<(NN * 32 + 255) / 256, 256, 0, stream>>>(x, Xp, NN);
    pad4to32_kernel<<<(256 * 32 + 255) / 256, 256, 0, stream>>>((const float*)d_in[CONV(0) + 0], Wq32, 256);
    pad4to32_kernel<<<(256 * 32 + 255) / 256, 256, 0, stream>>>((const float*)d_in[CONV(0) + 2], Wk32, 256);
    pad4to32_kernel<<<(256 * 32 + 255) / 256, 256, 0, stream>>>((const float*)d_in[CONV(0) + 4], Wv32, 256);
    pad4to32_kernel<<<(64 * 32 + 255) / 256, 256, 0, stream>>>((const float*)d_in[CONV(0) + 6], Ws32, 64);

    // ---- 4 TransformerConv layers ----
    for (int L = 0; L < 4; ++L) {
        const float* hin = (L == 0) ? Xp : ((L & 1) ? hA : hB);
        float*       ho  = (L & 1) ? hB : hA;
        int Kin = (L == 0) ? 32 : DD;
        int p = CONV(L);
        const float* qW = (L == 0) ? Wq32 : (const float*)d_in[p + 0];
        const float* kW = (L == 0) ? Wk32 : (const float*)d_in[p + 2];
        const float* vW = (L == 0) ? Wv32 : (const float*)d_in[p + 4];
        const float* sW = (L == 0) ? Ws32 : (const float*)d_in[p + 6];
        const float* qb  = (const float*)d_in[p + 1];
        const float* kbi = (const float*)d_in[p + 3];
        const float* vb  = (const float*)d_in[p + 5];
        const float* sb  = (const float*)d_in[p + 7];
        const float* bW  = (const float*)d_in[p + 8];

        gemm(stream, hin, Kin, qW, qb,  Q,  HH * DD, NN, HH * DD, Kin, 0);
        gemm(stream, hin, Kin, kW, kbi, Kb, HH * DD, NN, HH * DD, Kin, 0);
        gemm(stream, hin, Kin, vW, vb,  V,  HH * DD, NN, HH * DD, Kin, 0);
        gemm(stream, hin, Kin, sW, sb,  S,  DD,      NN, DD,      Kin, 0);

        hipMemsetAsync(MB,  0, sizeof(unsigned) * NN * HH, stream);
        hipMemsetAsync(SB,  0, sizeof(float) * NN * HH, stream);
        hipMemsetAsync(OUT, 0, sizeof(float) * NN * DD, stream);

        edge_logits_kernel<<<(EE * HH + 255) / 256, 256, 0, stream>>>(Q, Kb, src, dst, AL, MB);
        edge_exp_kernel<<<(EE * HH + 255) / 256, 256, 0, stream>>>(AL, MB, dst, SB);
        edge_aggr_kernel<<<(EE * DD + 255) / 256, 256, 0, stream>>>(AL, SB, V, src, dst, OUT);
        node_epilogue_kernel<<<NN / 4, 256, 0, stream>>>(OUT, S, bW, ho);

        if (L < 3) {
            const float* bnw = (const float*)d_in[40 + 2 * L];
            const float* bnb = (const float*)d_in[41 + 2 * L];
            bn_stats_kernel<<<DD, 256, 0, stream>>>(ho, ST);
            bn_apply_kernel<<<(NN * DD + 255) / 256, 256, 0, stream>>>(ho, ST, bnw, bnb);
        }
    }
    const float* h4 = hB;   // layer-4 output

    // ---- Set2Set pooling (4 steps) ----
    const float* Wih = (const float*)d_in[46];
    const float* Whh = (const float*)d_in[47];
    const float* bih = (const float*)d_in[48];
    const float* bhh = (const float*)d_in[49];
    hipMemsetAsync(qstar, 0, sizeof(float) * BB * 128, stream);
    hipMemsetAsync(hh,    0, sizeof(float) * BB * DD, stream);
    hipMemsetAsync(cc,    0, sizeof(float) * BB * DD, stream);
    for (int it = 0; it < 4; ++it) {
        gemm(stream, qstar, 128, Wih, bih, gates, 256, BB, 256, 128, 0);
        gemm(stream, hh,    DD,  Whh, bhh, gates, 256, BB, 256, DD,  1);   // accumulate
        lstm_pointwise_kernel<<<(BB * DD + 255) / 256, 256, 0, stream>>>(gates, hh, cc, qstar);
        s2s_attn_kernel<<<(BB + 255) / 256, 256, 0, stream>>>(h4, hh, qstar);
    }

    // ---- head: inter + shear concat, mu/logvar, decoder ----
    gemm(stream, qstar, 128, (const float*)d_in[50], (const float*)d_in[51],
         gbuf, 128, BB, 127, 128, 0);
    shear_fix_kernel<<<(BB + 255) / 256, 256, 0, stream>>>(gbuf, shear);

    float* mu = out + (size_t)BB * (EDGE_OUT + ATOM_OUT);
    float* lv = mu  + (size_t)BB * 128;
    gemm(stream, gbuf, 128, (const float*)d_in[52], (const float*)d_in[53], mu, 128, BB, 128, 128, 0);
    gemm(stream, gbuf, 128, (const float*)d_in[54], (const float*)d_in[55], lv, 128, BB, 128, 128, 0);

    gemm(stream, mu, 128, (const float*)d_in[56], (const float*)d_in[57], d1, 512, BB, 512, 128, 2);
    gemm(stream, d1, 512, (const float*)d_in[58], (const float*)d_in[59], d2, 512, BB, 512, 512, 2);
    gemm(stream, d2, 512, (const float*)d_in[60], (const float*)d_in[61],
         out + (size_t)BB * EDGE_OUT, ATOM_OUT, BB, ATOM_OUT, 512, 0);
    gemm(stream, d2, 512, (const float*)d_in[62], (const float*)d_in[63],
         out, EDGE_OUT, BB, EDGE_OUT, 512, 0);
}
